// MultiHeadAttention_23038204576075
// MI455X (gfx1250) — compile-verified, matrix operand order checked by reading, not run on hardware
//
#include <hip/hip_runtime.h>

// MHA forward on gfx1250 (MI455X), wave32 + V_WMMA_F32_16X16X32_F16 + TDM.
// Compute-bound (~69 GFLOP vs ~50MB HBM traffic) -> all MACs through WMMA;
// attention K/V tiles staged by the Tensor Data Mover (tensor_load_to_lds).

#define D_MODEL 1024
#define N_HEADS 16
#define D_K     64
#define B_SZ    2
#define T_SEQ   2048

typedef __attribute__((ext_vector_type(16))) _Float16 v16h;
typedef __attribute__((ext_vector_type(8)))  _Float16 v8h;
typedef __attribute__((ext_vector_type(4)))  _Float16 v4h;
typedef __attribute__((ext_vector_type(8)))  float    v8f;
typedef __attribute__((ext_vector_type(4)))  float    v4f;
typedef __attribute__((ext_vector_type(4)))  unsigned int u32x4;
typedef __attribute__((ext_vector_type(8)))  int      i32x8;
typedef __attribute__((ext_vector_type(4)))  int      i32x4;

union Frag16 { v16h v; v8h h[2]; };

__device__ __forceinline__ v8f wmma_f16(const Frag16& a, const Frag16& b, v8f c) {
  // D = A(16x32 f16) * B(32x16 f16) + C(16x16 f32)
  return __builtin_amdgcn_wmma_f32_16x16x32_f16(
      /*neg_a=*/false, a.v, /*neg_b=*/false, b.v,
      /*c_mod=*/(short)0, c, /*reuse_a=*/false, /*reuse_b=*/false);
}

// TDM: DMA a 32x64 f16 tile (rows contiguous, row stride 64 elems) into LDS,
// padding each 128B row with 16B so LDS rows land on a 72-half stride.
// D# group0: count=1 | lds_addr | 57b global addr | type=2 (bits 127:126).
// D# group1: data_size=2B, pad_enable, pad_interval=32 DW (code 4),
//            pad_amount=4 DW (code 3), tensor 64x2048, tile 64x32, stride0=64.
__device__ __forceinline__ void tdm_load_tile_32x64(const _Float16* gptr,
                                                    unsigned int lds_off) {
  const unsigned long long ga = (unsigned long long)(size_t)gptr;
  u32x4 g0;
  g0[0] = 1u;                                   // count=1, user mode
  g0[1] = lds_off;                              // LDS byte address
  g0[2] = (unsigned int)ga;                     // global addr [31:0]
  g0[3] = ((unsigned int)(ga >> 32) & 0x01FFFFFFu) | 0x80000000u; // [56:32]|type=2
  i32x8 g1;
  g1[0] = (int)((1u << 16) | (1u << 20) | (4u << 22) | (3u << 25));
  g1[1] = (int)(64u << 16);                     // tensor_dim0 = 64 (lo16 @ b79:64->w1[31:16])
  g1[2] = (int)(2048u << 16);                   // dim0 hi=0 | tensor_dim1 lo16 = 2048
  g1[3] = (int)(64u << 16);                     // dim1 hi=0 | tile_dim0 = 64
  g1[4] = (int)32u;                             // tile_dim1 = 32 | tile_dim2 = 0
  g1[5] = (int)64;                              // tensor_dim0_stride lo32 = 64
  g1[6] = 0;                                    // stride0 hi | stride1 lo (unused, 2D)
  g1[7] = 0;
  const i32x4 z4 = {};                          // groups 2/3: zero (<=2D tensor)
  const i32x8 z8 = {};
  // 6-arg form (clang-23 / therock-10.0 headers): (g0, g1, g2, g3, g4, cpol)
  __builtin_amdgcn_tensor_load_to_lds(g0, g1, z4, z4, z8, 0);
}

// ---------------------------------------------------------------------------
// Kernel 1: QKV projection.  C[4096,1024] = X f32 * W^T (W is [out,in]).
// Both A (X rows) and B (W rows) fragments read contiguous k-runs.
// Output written as f16 in [B,H,T,dk] layout.
// ---------------------------------------------------------------------------
__global__ void __launch_bounds__(128)
qkv_proj_kernel(const float* __restrict__ X,
                const float* __restrict__ Wq, const float* __restrict__ Wk,
                const float* __restrict__ Wv,
                _Float16* __restrict__ Qb, _Float16* __restrict__ Kb,
                _Float16* __restrict__ Vb)
{
  __shared__ _Float16 As[64][40];   // 64x32 tile, rows padded to 40 halfs (16B multiple)
  __shared__ _Float16 Bs[64][40];

  const int tid  = threadIdx.x;
  const int w    = tid >> 5;
  const int lane = tid & 31;
  const int lm   = lane & 15;
  const int lh   = lane >> 4;        // half-wave select
  const int kb   = lh * 8;           // A/B frag k-base per ISA layout
  const int m0   = blockIdx.y * 64;
  const int n0   = blockIdx.x * 64;

  const float* W  = (blockIdx.z == 0) ? Wq : ((blockIdx.z == 1) ? Wk : Wv);
  _Float16*  Out  = (blockIdx.z == 0) ? Qb : ((blockIdx.z == 1) ? Kb : Vb);

  v8f acc[4] = {};

  for (int kc = 0; kc < D_MODEL; kc += 32) {
    #pragma unroll
    for (int i = 0; i < 4; ++i) {           // stage 64x32 A and B, f32 -> f16, x4 vectors
      const int e = (tid + i * 128) * 4;
      const int r = e >> 5, c = e & 31;
      const v4f xa = *(const v4f*)&X[(size_t)(m0 + r) * D_MODEL + kc + c];
      const v4f xb = *(const v4f*)&W[(size_t)(n0 + r) * D_MODEL + kc + c];
      *(v4h*)&As[r][c] = __builtin_convertvector(xa, v4h);
      *(v4h*)&Bs[r][c] = __builtin_convertvector(xb, v4h);
    }
    __syncthreads();

    Frag16 af;
    af.h[0] = *(const v8h*)&As[w * 16 + lm][kb];
    af.h[1] = *(const v8h*)&As[w * 16 + lm][kb + 16];
    #pragma unroll
    for (int j = 0; j < 4; ++j) {
      Frag16 bf;
      bf.h[0] = *(const v8h*)&Bs[j * 16 + lm][kb];
      bf.h[1] = *(const v8h*)&Bs[j * 16 + lm][kb + 16];
      acc[j] = wmma_f16(af, bf, acc[j]);
    }
    __syncthreads();
  }

  // C-layout: lane = column n, VGPR r = row (lanes>=16 hold rows 8..15)
  #pragma unroll
  for (int j = 0; j < 4; ++j) {
    const int n = n0 + j * 16 + lm;
    const int h = n >> 6, d = n & 63;
    #pragma unroll
    for (int r = 0; r < 8; ++r) {
      const int m = m0 + w * 16 + lh * 8 + r;
      const int b = m >> 11, t = m & 2047;
      Out[((size_t)(b * N_HEADS + h) * T_SEQ + t) * D_K + d] = (_Float16)acc[j][r];
    }
  }
}

// ---------------------------------------------------------------------------
// Kernel 2: causal flash attention per (b,h).  Block = 64 q-rows (4 waves x 16),
// streamed over key tiles of 32 with online softmax.  K and V tiles staged by
// the Tensor Data Mover; V relaid out in LDS for the PV B-fragment.
// ---------------------------------------------------------------------------
__global__ void __launch_bounds__(128)
attn_kernel(const _Float16* __restrict__ Qb, const _Float16* __restrict__ Kb,
            const _Float16* __restrict__ Vb, _Float16* __restrict__ AO)
{
  __shared__ _Float16 Ks[32][72];       // 32 keys x 64 dk (TDM-padded rows)
  __shared__ _Float16 Vr[32][72];       // V row-major staging (TDM-padded rows)
  __shared__ _Float16 Vt[64][40];       // V transposed: [dk][key]
  __shared__ _Float16 Ps[4][16][40];    // per-wave P tile for C->A relayout

  const int tid  = threadIdx.x;
  const int w    = tid >> 5;
  const int lane = tid & 31;
  const int lm   = lane & 15;
  const int lh   = lane >> 4;
  const int kb   = lh * 8;
  const int qt   = blockIdx.x;          // 0..31  (q tile of 64)
  const int bh   = blockIdx.y;          // 0..31  (b*16+h)
  const int q0   = qt * 64;
  const int b    = bh >> 4, h = bh & 15;
  const size_t bhbase = (size_t)bh * T_SEQ * D_K;
  const float  sc = 0.125f;             // 1/sqrt(64)

  const unsigned int ks_off = (unsigned int)(size_t)&Ks[0][0];
  const unsigned int vr_off = (unsigned int)(size_t)&Vr[0][0];

  // Q A-frags (16 rows x 64 dk = two k-chunks), straight from global (contiguous)
  Frag16 qf[2];
  {
    const _Float16* qrow = Qb + bhbase + (size_t)(q0 + w * 16 + lm) * D_K;
    #pragma unroll
    for (int c = 0; c < 2; ++c) {
      qf[c].h[0] = *(const v8h*)(qrow + c * 32 + kb);
      qf[c].h[1] = *(const v8h*)(qrow + c * 32 + kb + 16);
    }
  }

  float mrow[8], lrow[8];
  v8f o[4] = {};
  #pragma unroll
  for (int r = 0; r < 8; ++r) { mrow[r] = -__builtin_inff(); lrow[r] = 0.0f; }

  const int nkt = 2 * (qt + 1);         // causal: last key tile covers q0+63
  for (int kt = 0; kt < nkt; ++kt) {
    const int k0 = kt * 32;

    if (w == 0) {                       // wave-uniform branch: one TDM issue per tile
      tdm_load_tile_32x64(Kb + bhbase + (size_t)k0 * D_K, ks_off);
      tdm_load_tile_32x64(Vb + bhbase + (size_t)k0 * D_K, vr_off);
      __builtin_amdgcn_s_wait_tensorcnt(0);
    }
    __syncthreads();                    // Ks / Vr visible to all waves

    #pragma unroll
    for (int i = 0; i < 16; ++i) {      // LDS->LDS transpose of V tile
      const int e = tid + i * 128;
      const int d = e & 63, kk = e >> 6;
      Vt[d][kk] = Vr[kk][d];
    }
    __syncthreads();                    // Vt ready

    // S = Q K^T : two 16x16 blocks (keys 0-15, 16-31), chained over dk chunks
    v8f s0 = {}, s1 = {};
    #pragma unroll
    for (int c = 0; c < 2; ++c) {
      Frag16 bf;
      bf.h[0] = *(const v8h*)&Ks[lm][c * 32 + kb];
      bf.h[1] = *(const v8h*)&Ks[lm][c * 32 + kb + 16];
      s0 = wmma_f16(qf[c], bf, s0);
      bf.h[0] = *(const v8h*)&Ks[16 + lm][c * 32 + kb];
      bf.h[1] = *(const v8h*)&Ks[16 + lm][c * 32 + kb + 16];
      s1 = wmma_f16(qf[c], bf, s1);
    }

    // causal mask + online softmax (row stats via 16-lane xor reductions)
    const int kg0 = k0 + lm, kg1 = k0 + 16 + lm;
    float scalef[8], p0[8], p1[8];
    #pragma unroll
    for (int r = 0; r < 8; ++r) {
      const int qg = q0 + w * 16 + lh * 8 + r;
      const float a0 = (kg0 <= qg) ? s0[r] * sc : -__builtin_inff();
      const float a1 = (kg1 <= qg) ? s1[r] * sc : -__builtin_inff();
      float cm = fmaxf(a0, a1);
      cm = fmaxf(cm, __shfl_xor(cm, 1));
      cm = fmaxf(cm, __shfl_xor(cm, 2));
      cm = fmaxf(cm, __shfl_xor(cm, 4));
      cm = fmaxf(cm, __shfl_xor(cm, 8));
      const float mnew = fmaxf(mrow[r], cm);
      const float f = __expf(mrow[r] - mnew);   // 0 on first tile (m=-inf)
      mrow[r] = mnew; scalef[r] = f;
      p0[r] = __expf(a0 - mnew);
      p1[r] = __expf(a1 - mnew);
      float rs = p0[r] + p1[r];
      rs += __shfl_xor(rs, 1);
      rs += __shfl_xor(rs, 2);
      rs += __shfl_xor(rs, 4);
      rs += __shfl_xor(rs, 8);
      lrow[r] = lrow[r] * f + rs;
    }

    // P: C-layout -> LDS -> A-layout (same-wave LDS ordering, no barrier needed)
    #pragma unroll
    for (int r = 0; r < 8; ++r) {
      Ps[w][lh * 8 + r][lm]      = (_Float16)p0[r];
      Ps[w][lh * 8 + r][16 + lm] = (_Float16)p1[r];
    }
    Frag16 pf;
    pf.h[0] = *(const v8h*)&Ps[w][lm][kb];
    pf.h[1] = *(const v8h*)&Ps[w][lm][kb + 16];

    // O = O*diag(scale) + P V   (V transposed in LDS -> contiguous B-frag)
    #pragma unroll
    for (int j = 0; j < 4; ++j) {
      Frag16 vf;
      vf.h[0] = *(const v8h*)&Vt[j * 16 + lm][kb];
      vf.h[1] = *(const v8h*)&Vt[j * 16 + lm][kb + 16];
      #pragma unroll
      for (int r = 0; r < 8; ++r) o[j][r] *= scalef[r];
      o[j] = wmma_f16(pf, vf, o[j]);
    }
    __syncthreads();                    // all compute done before next TDM overwrite
  }

  // normalize and write [B,T,D_MODEL] f16 activations for the output projection
  const size_t obase = (size_t)b * T_SEQ * D_MODEL;
  #pragma unroll
  for (int j = 0; j < 4; ++j) {
    #pragma unroll
    for (int r = 0; r < 8; ++r) {
      const int t = q0 + w * 16 + lh * 8 + r;
      const int dcol = h * 64 + j * 16 + lm;
      AO[obase + (size_t)t * D_MODEL + dcol] = (_Float16)(o[j][r] / lrow[r]);
    }
  }
}

// ---------------------------------------------------------------------------
// Kernel 3: output projection.  out f32 = AO f16 * Wo^T.
// ---------------------------------------------------------------------------
__global__ void __launch_bounds__(128)
out_proj_kernel(const _Float16* __restrict__ A, const float* __restrict__ Wo,
                float* __restrict__ Out)
{
  __shared__ _Float16 As[64][40];
  __shared__ _Float16 Bs[64][40];

  const int tid  = threadIdx.x;
  const int w    = tid >> 5;
  const int lane = tid & 31;
  const int lm   = lane & 15;
  const int lh   = lane >> 4;
  const int kb   = lh * 8;
  const int m0   = blockIdx.y * 64;
  const int n0   = blockIdx.x * 64;

  v8f acc[4] = {};

  for (int kc = 0; kc < D_MODEL; kc += 32) {
    #pragma unroll
    for (int i = 0; i < 2; ++i) {           // A is already f16: 8-half vector copies
      const int e = (tid + i * 128) * 8;
      const int r = e >> 5, c = e & 31;
      *(v8h*)&As[r][c] = *(const v8h*)&A[(size_t)(m0 + r) * D_MODEL + kc + c];
    }
    #pragma unroll
    for (int i = 0; i < 4; ++i) {           // Wo f32 -> f16 staged as x4 vectors
      const int e = (tid + i * 128) * 4;
      const int r = e >> 5, c = e & 31;
      const v4f xb = *(const v4f*)&Wo[(size_t)(n0 + r) * D_MODEL + kc + c];
      *(v4h*)&Bs[r][c] = __builtin_convertvector(xb, v4h);
    }
    __syncthreads();

    Frag16 af;
    af.h[0] = *(const v8h*)&As[w * 16 + lm][kb];
    af.h[1] = *(const v8h*)&As[w * 16 + lm][kb + 16];
    #pragma unroll
    for (int j = 0; j < 4; ++j) {
      Frag16 bf;
      bf.h[0] = *(const v8h*)&Bs[j * 16 + lm][kb];
      bf.h[1] = *(const v8h*)&Bs[j * 16 + lm][kb + 16];
      acc[j] = wmma_f16(af, bf, acc[j]);
    }
    __syncthreads();
  }

  #pragma unroll
  for (int j = 0; j < 4; ++j) {
    const int n = n0 + j * 16 + lm;
    #pragma unroll
    for (int r = 0; r < 8; ++r) {
      const int m = m0 + w * 16 + lh * 8 + r;
      Out[(size_t)m * D_MODEL + n] = acc[j][r];
    }
  }
}

// ---------------------------------------------------------------------------
extern "C" void kernel_launch(void* const* d_in, const int* in_sizes, int n_in,
                              void* d_out, int out_size, void* d_ws, size_t ws_size,
                              hipStream_t stream) {
  const float* x  = (const float*)d_in[0];
  // d_in[1] = causal mask (tril) -- implemented analytically, not read.
  const float* Wq = (const float*)d_in[2];
  const float* Wk = (const float*)d_in[3];
  const float* Wv = (const float*)d_in[4];
  const float* Wo = (const float*)d_in[5];

  // Workspace: Q,K,V in [B,H,T,dk] f16 (8MB each) + AO [B,T,D] f16 (8MB) = 32MB
  const size_t NE = (size_t)B_SZ * T_SEQ * D_MODEL;   // 4,194,304 elements
  _Float16* Qb = (_Float16*)d_ws;
  _Float16* Kb = Qb + NE;
  _Float16* Vb = Kb + NE;
  _Float16* AO = Vb + NE;

  qkv_proj_kernel<<<dim3(16, 64, 3), 128, 0, stream>>>(x, Wq, Wk, Wv, Qb, Kb, Vb);
  attn_kernel   <<<dim3(32, 32),     128, 0, stream>>>(Qb, Kb, Vb, AO);
  out_proj_kernel<<<dim3(16, 64),    128, 0, stream>>>(AO, Wo, (float*)d_out);
}
